// EncoderDecoderConvLSTM_6064493822082
// MI455X (gfx1250) — matrix operand order repaired and checked by reading, not compiled, not run on hardware
//
#include <hip/hip_runtime.h>
#include <hip/hip_bf16.h>

typedef __attribute__((ext_vector_type(16))) __bf16        v16bf;
typedef __attribute__((ext_vector_type(8)))  float         v8f;

union BFrag { uint4 q[2]; v16bf b; };

static __device__ __forceinline__ unsigned short f2bf(float f) {
    union { float f; unsigned int u; } t; t.f = f;
    unsigned int u = t.u;
    u += 0x7fffu + ((u >> 16) & 1u);   // round-to-nearest-even
    return (unsigned short)(u >> 16);
}
static __device__ __forceinline__ float bf2f(unsigned short v) {
    union { unsigned int u; float f; } t; t.u = ((unsigned int)v) << 16; return t.f;
}
static __device__ __forceinline__ float fast_sigmoid(float x) {
    return 1.f / (1.f + __expf(-x));
}
static __device__ __forceinline__ float fast_tanh(float x) {
    return 1.f - 2.f / (__expf(2.f * x) + 1.f);   // stays on v_exp_f32
}

// ---------------------------------------------------------------------------
// Pack conv weights W[n][c][ky][kx] (fp32) into a contiguous-K WMMA bf16
// B-operand layout: chunks of 16 bf16 per (ktile, ntile, lane).
// B lane l holds N = l%16; half = l/16; VGPR pair p holds K = 2p + 16*half (+j).
// ---------------------------------------------------------------------------
template<int CK, int KREAL, int KPAD>
__global__ void pack_weights(const float* __restrict__ W, unsigned short* __restrict__ out) {
    const int total = KPAD * 256;
    for (int o = blockIdx.x * blockDim.x + threadIdx.x; o < total; o += gridDim.x * blockDim.x) {
        int e     = o & 15;
        int chunk = o >> 4;
        int lane  = chunk & 31;
        int nt    = (chunk >> 5) & 15;
        int kt    = chunk >> 9;
        int half  = lane >> 4, nl = lane & 15;
        int p = e >> 1, j = e & 1;
        int K = kt * 32 + 2 * p + 16 * half + j;
        int n = nt * 16 + nl;
        float v = 0.f;
        if (K < KREAL) v = W[(n * CK + (K / 9)) * 9 + (K % 9)];
        out[o] = f2bf(v);
    }
}

// x_t (fp32, strided [B,12,1,64,64]) -> channel 0 of the enc comb buffer (bf16)
__global__ void xcopy(const float* __restrict__ x, unsigned short* __restrict__ dst, int t) {
    int i = blockIdx.x * 256 + threadIdx.x;          // B*4096
    int b = i >> 12, s = i & 4095;
    dst[(long)b * 65 * 4096 + s] = f2bf(x[((long)b * 12 + t) * 4096 + s]);
}

// encoder final h (enc comb ch 1..64) -> encoder_vector output (fp32)
//                                      + decoder comb t=0 first half (bf16)
__global__ void transition(const unsigned short* __restrict__ combE0,
                           float* __restrict__ evout,
                           unsigned short* __restrict__ combD0) {
    for (long i = blockIdx.x * 256l + threadIdx.x; i < 32l * 64 * 4096; i += (long)gridDim.x * 256) {
        long b = i >> 18;              // 64*4096
        long r = i & 262143;           // ch*4096 + s
        unsigned short v = combE0[b * 65 * 4096 + 4096 + r];
        evout[i] = bf2f(v);
        combD0[b * 128 * 4096 + r] = v;
    }
}

// ---------------------------------------------------------------------------
// One ConvLSTM step as implicit GEMM on WMMA bf16, gates fused.
//   comb: [B][CC][64][64] bf16 combined input (CC = 65 enc, 128 dec)
//   Block: 32 consecutive x positions, fixed (b,y)  => raw 3x34xCC halo tile
//   is staged to LDS ONCE; per-K-step im2col is pure LDS (no global, no
//   predication). 8 waves: M-tile (w&1) x gate group (w>>1), 4 WMMAs each.
// ---------------------------------------------------------------------------
template<int CC, int KREAL, int KPAD>
__global__ __launch_bounds__(256)
void convlstm_step(const unsigned short* __restrict__ comb,
                   const unsigned short* __restrict__ wpack,
                   const float* __restrict__ bias,
                   float* __restrict__ cstate,
                   unsigned short* __restrict__ hout, long out_bs,
                   unsigned short* __restrict__ hout2)
{
    constexpr int KT  = KPAD / 32;
    constexpr int RAW = CC * 102;                       // CC channels * 3 rows * 34 cols
    __shared__ __align__(16) unsigned int Alds[32 * 16]; // 32 m-rows x 16 K-pairs
    __shared__ float gbuf[32 * 256];                     // gates, fp32
    __shared__ unsigned int Dpair[KPAD / 2];             // K-pair -> raw displacements
    __shared__ unsigned short Raw[RAW + 34];             // halo tile + zero pad region

    const int tid   = threadIdx.x;
    const int lane  = tid & 31;            // wave32
    const int w     = tid >> 5;
    const int mtile = w & 1;
    const int ngrp  = w >> 1;              // gate 0..3 (i,f,o,g)
    const int half  = lane >> 4;
    const int nl    = lane & 15;
    const int pbase = blockIdx.x * 32;
    const int b     = pbase >> 12;         // uniform across block
    const int y     = (pbase >> 6) & 63;   // uniform
    const int x0    = pbase & 63;          // uniform (0 or 32)

    // ---- one-time displacement table: K -> (c*3+dy)*34 + dx ; pad -> zero region
    for (int q = tid; q < KPAD / 2; q += 256) {
        unsigned d[2];
        #pragma unroll
        for (int j = 0; j < 2; ++j) {
            int K = 2 * q + j;
            if (K < KREAL) { int c = K / 9, r = K % 9; d[j] = (unsigned)((c * 3 + r / 3) * 34 + r % 3); }
            else           d[j] = (unsigned)RAW;
        }
        Dpair[q] = d[0] | (d[1] << 16);
    }
    if (tid < 34) Raw[RAW + tid] = 0;

    // ---- one-time raw halo tile load (zero-padded borders) ----
    {
        const unsigned short* cb = comb + (long)b * CC * 4096;
        for (int idx = tid; idx < RAW; idx += 256) {
            int c   = idx / 102;
            int rr  = idx - c * 102;
            int dy  = rr / 34;
            int col = rr - dy * 34;
            int yy  = y + dy - 1;
            int xx  = x0 + col - 1;
            unsigned short v = 0;
            if ((unsigned)(yy | xx) < 64u) v = cb[(c << 12) + (yy << 6) + xx];
            Raw[idx] = v;
        }
    }

    // per-thread staging coords (loop-invariant)
    int skp[2], smm[2];
    #pragma unroll
    for (int i = 0; i < 2; ++i) {
        int idx = tid + i * 256;
        smm[i] = idx >> 4;      // m position 0..31
        skp[i] = idx & 15;      // K-pair slot 0..15
    }

    v8f acc[4] = {};

    for (int kt = 0; kt < KT; ++kt) {
        __syncthreads();
        // ---- im2col A tile from LDS raw tile: no global, no predication ----
        #pragma unroll
        for (int i = 0; i < 2; ++i) {
            unsigned int dd = Dpair[kt * 16 + skp[i]];
            unsigned int lo = Raw[(dd & 0xffffu) + smm[i]];
            unsigned int hi = Raw[(dd >> 16)     + smm[i]];
            Alds[tid + i * 256] = lo | (hi << 16);
        }
        __syncthreads();

        // ---- A fragment (contiguous K = 16h + s, as packed B): two ds_load_b128 ----
        const unsigned int* arow = &Alds[(mtile * 16 + nl) * 16 + half * 8];
        BFrag ua;
        ua.q[0] = *(const uint4*)(arow);        // K pairs 8h..8h+3   -> K 0..7 / 16..23
        ua.q[1] = *(const uint4*)(arow + 4);    // K pairs 8h+4..8h+7 -> K 8..15 / 24..31
        v16bf afrag = ua.b;

        // ---- B fragments: pre-packed, two global_load_b128 each ----
        v16bf bfrag[4];
        #pragma unroll
        for (int nt = 0; nt < 4; ++nt) {
            int ntg = ngrp * 4 + nt;
            const unsigned short* bp = wpack + ((((long)kt * 16 + ntg) * 32 + lane) << 4);
            BFrag ub;
            ub.q[0] = *(const uint4*)(bp);
            ub.q[1] = *(const uint4*)(bp + 8);
            bfrag[nt] = ub.b;
        }
        if (kt + 1 < KT)
            __builtin_prefetch(wpack + ((((long)(kt + 1) * 16 + ngrp * 4) * 32 + lane) << 4), 0, 1);

        // ---- 4 x v_wmma_f32_16x16x32_bf16 ----
        #pragma unroll
        for (int nt = 0; nt < 4; ++nt)
            acc[nt] = __builtin_amdgcn_wmma_f32_16x16x32_bf16(
                false, afrag, false, bfrag[nt], (short)0, acc[nt], false, false);
    }

    // ---- epilogue: bias + scatter C fragments to LDS gate buffer ----
    // C/D layout: lane 0-15 -> M=r, N=lane ; lane 16-31 -> M=r+8, N=lane-16
    #pragma unroll
    for (int nt = 0; nt < 4; ++nt) {
        int n = (ngrp * 4 + nt) * 16 + nl;
        float bv = bias[n];
        #pragma unroll
        for (int r = 0; r < 8; ++r) {
            int m = mtile * 16 + r + half * 8;
            gbuf[m * 256 + n] = acc[nt][r] + bv;
        }
    }
    __syncthreads();

    // ---- fused LSTM pointwise: c' = f*c + i*g ; h' = o*tanh(c') ----
    for (int idx = tid; idx < 2048; idx += 256) {
        int m = idx >> 6, ch = idx & 63;
        float ci = gbuf[m * 256 + ch];
        float cf = gbuf[m * 256 + 64 + ch];
        float co = gbuf[m * 256 + 128 + ch];
        float cg = gbuf[m * 256 + 192 + ch];
        float gi = fast_sigmoid(ci);
        float gf = fast_sigmoid(cf);
        float go = fast_sigmoid(co);
        float gg = fast_tanh(cg);
        long sp  = (y << 6) + x0 + m;
        long cix = (((long)b * 64 + ch) << 12) + sp;
        float cn = gf * cstate[cix] + gi * gg;
        cstate[cix] = cn;
        unsigned short hv = f2bf(go * fast_tanh(cn));
        long ho = (long)b * out_bs + ((long)ch << 12) + sp;
        hout[ho] = hv;
        if (hout2) hout2[ho] = hv;   // decoder: duplicate into second comb half
    }
}

// ---------------------------------------------------------------------------
// (1,3,3) 3D conv head == per-frame 2D conv, 64->1 channels. Negligible FLOPs.
// ---------------------------------------------------------------------------
__global__ __launch_bounds__(256)
void head_conv(const unsigned short* __restrict__ h, long bstride,
               const float* __restrict__ Wc, const float* __restrict__ bc,
               float* __restrict__ yout, int t)
{
    __shared__ float wk[576];
    for (int i = threadIdx.x; i < 576; i += 256) wk[i] = Wc[i];
    __syncthreads();
    int p = blockIdx.x * 256 + threadIdx.x;
    int b = p >> 12, y = (p >> 6) & 63, x = p & 63;
    float s = bc[0];
    for (int c = 0; c < 64; ++c) {
        const unsigned short* hc = h + (long)b * bstride + ((long)c << 12);
        #pragma unroll
        for (int ky = 0; ky < 3; ++ky) {
            int yy = y + ky - 1;
            if ((unsigned)yy >= 64u) continue;
            #pragma unroll
            for (int kx = 0; kx < 3; ++kx) {
                int xx = x + kx - 1;
                if ((unsigned)xx >= 64u) continue;
                s += bf2f(hc[(yy << 6) + xx]) * wk[(c * 3 + ky) * 3 + kx];
            }
        }
    }
    yout[(((long)b * 12 + t) << 12) + (y << 6) + x] = s;
}

// ---------------------------------------------------------------------------
static inline size_t align256(size_t x) { return (x + 255) & ~size_t(255); }

extern "C" void kernel_launch(void* const* d_in, const int* in_sizes, int n_in,
                              void* d_out, int out_size, void* d_ws, size_t ws_size,
                              hipStream_t stream) {
    (void)in_sizes; (void)n_in; (void)out_size; (void)ws_size;
    const float* x     = (const float*)d_in[0];
    /* d_in[1] = future_step (==12, fixed by reference) */
    const float* W_enc = (const float*)d_in[2];
    const float* b_enc = (const float*)d_in[3];
    const float* W_dec = (const float*)d_in[4];
    const float* b_dec = (const float*)d_in[5];
    const float* W_cnn = (const float*)d_in[6];
    const float* b_cnn = (const float*)d_in[7];

    float* yout  = (float*)d_out;                 // [32,12,1,64,64]
    float* evout = yout + (long)32 * 12 * 4096;   // [32, 64*64*64]

    // ---- workspace carve ----
    const size_t CE = (size_t)32 * 65  * 4096;    // enc comb elements
    const size_t CD = (size_t)32 * 128 * 4096;    // dec comb elements
    const size_t HS = (size_t)32 * 64  * 4096;    // h/c elements
    char* base = (char*)d_ws;
    size_t off = 0;
    unsigned short* wpe    = (unsigned short*)(base + off); off = align256(off + (size_t)608  * 256 * 2);
    unsigned short* wpd    = (unsigned short*)(base + off); off = align256(off + (size_t)1152 * 256 * 2);
    unsigned short* combE0 = (unsigned short*)(base + off); off = align256(off + CE * 2);
    unsigned short* combE1 = (unsigned short*)(base + off); off = align256(off + CE * 2);
    unsigned short* combD0 = (unsigned short*)(base + off); off = align256(off + CD * 2);
    unsigned short* combD1 = (unsigned short*)(base + off); off = align256(off + CD * 2);
    float*          cst    = (float*)(base + off);          off = align256(off + HS * 4);

    // ---- prep (deterministic, redone every call) ----
    pack_weights<65, 585, 608>  <<<304, 256, 0, stream>>>(W_enc, wpe);
    pack_weights<128,1152,1152> <<<576, 256, 0, stream>>>(W_dec, wpd);
    hipMemsetAsync(combE0, 0, CE * 2, stream);    // h0 = 0 (x ch filled by xcopy)
    hipMemsetAsync(combD0, 0, CD * 2, stream);    // dec t=0 second half = h0 = 0
    hipMemsetAsync(cst,    0, HS * 4, stream);    // c0 = 0

    // ---- encoder: 12 steps, comb = [x_t | h] ----
    unsigned short* cE[2] = { combE0, combE1 };
    for (int t = 0; t < 12; ++t) {
        xcopy<<<512, 256, 0, stream>>>(x, cE[t & 1], t);
        convlstm_step<65, 585, 608><<<4096, 256, 0, stream>>>(
            cE[t & 1], wpe, b_enc, cst,
            cE[(t + 1) & 1] + 4096, 65l * 4096, nullptr);   // h -> ch 1..64 of next comb
    }
    // final encoder h in combE0 ch1..64 -> encoder_vector out + decoder comb[0] first half
    transition<<<2048, 256, 0, stream>>>(combE0, evout, combD0);

    // ---- decoder: carry (ev,h,c)->(h,h,c): comb = [h_prev | h_prev] ----
    hipMemsetAsync(cst, 0, HS * 4, stream);
    unsigned short* cD[2] = { combD0, combD1 };
    for (int t = 0; t < 12; ++t) {
        unsigned short* nxt = cD[(t + 1) & 1];
        convlstm_step<128, 1152, 1152><<<4096, 256, 0, stream>>>(
            cD[t & 1], wpd, b_dec, cst,
            nxt, 128l * 4096, nxt + 64 * 4096);             // h duplicated into both halves
        head_conv<<<512, 256, 0, stream>>>(nxt, 128l * 4096, W_cnn, b_cnn, yout, t);
    }
}
